// HATGNN_15917148799304
// MI455X (gfx1250) — matrix operand order repaired and checked by reading, not compiled, not run on hardware
//
#include <hip/hip_runtime.h>
#include <hip/hip_bf16.h>

typedef __attribute__((ext_vector_type(16))) _Float16 v16h;
typedef __attribute__((ext_vector_type(8)))  float    v8f;

#define BATCH 8
#define PD 128
#define NN 2048          // nodes per sample after conv stem (16*128)

// ---------------------------------------------------------------------------
// order-preserving float <-> uint encoding for atomic max
__device__ __forceinline__ unsigned enc_f(float f) {
  unsigned u = __float_as_uint(f);
  return (u & 0x80000000u) ? ~u : (u | 0x80000000u);
}
__device__ __forceinline__ float dec_f(unsigned u) {
  return __uint_as_float((u & 0x80000000u) ? (u & 0x7fffffffu) : ~u);
}
#define ENC_NEGINF 0x007fffffu

__device__ __forceinline__ float gelu_exact(float v) {
  return 0.5f * v * (1.0f + erff(v * 0.70710678118654752f));
}

__device__ __forceinline__ void pack8(v16h& d, int base, float4 lo, float4 hi) {
  d[base + 0] = (_Float16)lo.x; d[base + 1] = (_Float16)lo.y;
  d[base + 2] = (_Float16)lo.z; d[base + 3] = (_Float16)lo.w;
  d[base + 4] = (_Float16)hi.x; d[base + 5] = (_Float16)hi.y;
  d[base + 6] = (_Float16)hi.z; d[base + 7] = (_Float16)hi.w;
}

// ---------------------------------------------------------------------------
// WMMA GEMM:  C[bz, m, n] = sum_k A[bz][m][k] * W[bz][n][k] + bias[n]
// One wave per 16x64 output strip (4 accumulators reuse one A fragment).
// Requires M%16==0, N%64==0, KD%32==0, rows 16B-aligned (KD%4==0 floats).
// grid = (N/64, M/16, batch), block = 32 threads (1 wave).
template <int KD>
__global__ void gemm_xwt_wmma(const float* __restrict__ A, const float* __restrict__ Wm,
                              const float* __restrict__ bias, float* __restrict__ C,
                              int M, int N,
                              long long sA, long long sW, long long sC) {
  const int lane = threadIdx.x & 31;
  const int half = lane >> 4;
  const int l16  = lane & 15;
  const int tileN = blockIdx.x * 64;
  const int tileM = blockIdx.y * 16;
  const int bz = blockIdx.z;

  const float* Ab = A + (long long)bz * sA;
  const float* Wb = Wm + (long long)bz * sW;
  float*       Cb = C + (long long)bz * sC;

  // A 16x32 f16 fragment source: lane row = tileM + (lane&15);
  // K chunks [half*8, +8) and [half*8+16, +8)
  const float* arow = Ab + (size_t)(tileM + l16) * KD + half * 8;
  // B 32x16 f16 fragment source: lane column = (n-tile) + (lane&15);
  // K chunks [half*8, +8) and [half*8+16, +8)
  const float* wrow = Wb + (size_t)(tileN + l16) * KD + half * 8;

  v8f acc[4] = {v8f{}, v8f{}, v8f{}, v8f{}};

#pragma unroll 4
  for (int kk = 0; kk < KD; kk += 32) {
    v16h a;
    {
      const float4* p0 = (const float4*)(arow + kk);
      const float4* p1 = (const float4*)(arow + kk + 16);
      pack8(a, 0, p0[0], p0[1]);
      pack8(a, 8, p1[0], p1[1]);
    }
#pragma unroll
    for (int t = 0; t < 4; ++t) {
      const float4* wp = (const float4*)(wrow + (size_t)t * 16 * KD + kk);
      v16h b;
      pack8(b, 0, wp[0], wp[1]);
      pack8(b, 8, wp[4], wp[5]);
      acc[t] = __builtin_amdgcn_wmma_f32_16x16x32_f16(false, a, false, b, (short)0,
                                                      acc[t], false, false);
    }
  }

#pragma unroll
  for (int t = 0; t < 4; ++t) {
    const int n = tileN + t * 16 + l16;
    const float bv = bias ? bias[n] : 0.0f;
#pragma unroll
    for (int r = 0; r < 8; ++r) {
      const int m = tileM + r + half * 8;          // C layout: VGPR r, half -> M
      Cb[(size_t)m * N + n] = acc[t][r] + bv;
    }
  }
}

// ---------------------------------------------------------------------------
// direct conv3x3 + BN (inference) + exact GELU.  NCHW / OIHW, SAME padding.
__global__ void conv_bn_gelu_k(const float* __restrict__ x, const float* __restrict__ w,
                               const float* __restrict__ cb, const float* __restrict__ gs,
                               const float* __restrict__ bs, float* __restrict__ y,
                               int Bn, int Cin, int Hin, int Win,
                               int Cout, int Hout, int Wout, int stride, int pad) {
  size_t tid = blockIdx.x * (size_t)blockDim.x + threadIdx.x;
  size_t total = (size_t)Bn * Cout * Hout * Wout;
  if (tid >= total) return;
  int ow = tid % Wout; size_t r = tid / Wout;
  int oh = r % Hout;   r /= Hout;
  int co = r % Cout;   int b = (int)(r / Cout);
  float acc = 0.f;
  for (int ci = 0; ci < Cin; ++ci) {
    const float* xp = x + ((size_t)(b * Cin + ci) * Hin) * Win;
    const float* wp = w + (size_t)(co * Cin + ci) * 9;
#pragma unroll
    for (int kh = 0; kh < 3; ++kh) {
      int ih = oh * stride + kh - pad;
      if (ih < 0 || ih >= Hin) continue;
#pragma unroll
      for (int kw = 0; kw < 3; ++kw) {
        int iw = ow * stride + kw - pad;
        if (iw < 0 || iw >= Win) continue;
        acc += xp[(size_t)ih * Win + iw] * wp[kh * 3 + kw];
      }
    }
  }
  float v = (acc + cb[co]) * gs[co] + bs[co];
  y[tid] = gelu_exact(v);
}

// feat[b,n,c] = conv4[b,c,n] + pos_emb[n,c]
__global__ void feat_k(const float* __restrict__ c4, const float* __restrict__ pos,
                       float* __restrict__ x) {
  size_t tid = blockIdx.x * (size_t)blockDim.x + threadIdx.x;
  if (tid >= (size_t)BATCH * NN * PD) return;
  int c = tid % PD; size_t r = tid / PD;
  int n = r % NN;   int b = (int)(r / NN);
  x[tid] = c4[((size_t)b * PD + c) * NN + n] + pos[(size_t)n * PD + c];
}

__global__ void diag_k(const float* __restrict__ G, float* __restrict__ sq) {
  int tid = blockIdx.x * blockDim.x + threadIdx.x;
  if (tid >= BATCH * NN) return;
  int b = tid / NN, i = tid % NN;
  sq[tid] = G[(size_t)b * NN * NN + (size_t)i * NN + i];
}

// per-node scan of Gram row: keep kd smallest distances, subsample by dilation
__global__ void knn_k(const float* __restrict__ G, const float* __restrict__ sq,
                      int* __restrict__ nbr, int kd, int dil) {
  int tid = blockIdx.x * blockDim.x + threadIdx.x;
  if (tid >= BATCH * NN) return;
  int b = tid / NN, i = tid % NN;
  const float* Gr  = G + ((size_t)b * NN + i) * NN;
  const float* sqb = sq + (size_t)b * NN;
  float bd[18]; int bi[18];
  for (int t = 0; t < kd; ++t) { bd[t] = 3.4e38f; bi[t] = 0; }
  const float si = sqb[i];
  for (int j = 0; j < NN; ++j) {
    float d = si + sqb[j] - 2.0f * Gr[j];
    if (j == i) d += 1e10f;                       // loop=False
    if (d < bd[kd - 1]) {
      int pos = kd - 1;
      while (pos > 0 && d < bd[pos - 1]) {
        bd[pos] = bd[pos - 1]; bi[pos] = bi[pos - 1]; --pos;
      }
      bd[pos] = d; bi[pos] = j;
    }
  }
  int* out = nbr + ((size_t)b * NN + i) * 9;
  for (int m = 0; m < 9; ++m) out[m] = bi[m * dil];
}

__global__ void fill_u32_k(unsigned* __restrict__ p, unsigned val, size_t n) {
  size_t tid = blockIdx.x * (size_t)blockDim.x + threadIdx.x;
  if (tid < n) p[tid] = val;
}

// scatter-max: md[dst] = max(h[src]-h[dst]) over edges, via monotone-uint atomicMax
__global__ void edge_max_k(const float* __restrict__ h, const int* __restrict__ nbr,
                           unsigned* __restrict__ md) {
  size_t tid = blockIdx.x * (size_t)blockDim.x + threadIdx.x;
  if (tid >= (size_t)BATCH * NN * 9 * PD) return;
  int c = tid & 127; size_t r = tid >> 7;
  int m = r % 9;     r /= 9;
  int i = r % NN;    int b = (int)(r / NN);
  int j = nbr[((size_t)b * NN + i) * 9 + m];
  float v = h[((size_t)b * NN + i) * PD + c] - h[((size_t)b * NN + j) * PD + c];
  atomicMax(&md[((size_t)b * NN + j) * PD + c], enc_f(v));
}

// cat[b,i,:] = [h[b,i,:], decode(md)[b,i,:] with -inf -> 0]
__global__ void concat_hmd_k(const float* __restrict__ h, const unsigned* __restrict__ md,
                             float* __restrict__ cat) {
  size_t tid = blockIdx.x * (size_t)blockDim.x + threadIdx.x;
  if (tid >= (size_t)BATCH * NN * PD) return;
  int c = tid % PD; size_t row = tid / PD;
  float mv = dec_f(md[tid]);
  if (__float_as_uint(mv) == 0xff800000u) mv = 0.0f;   // empty segment -> 0
  cat[row * (2 * PD) + c]      = h[tid];
  cat[row * (2 * PD) + PD + c] = mv;
}

// y = LayerNorm(x + r) * g + b   (r == nullptr means r = 0; y may alias x)
__global__ void residual_ln_k(const float* __restrict__ xin, const float* __restrict__ r,
                              const float* __restrict__ g, const float* __restrict__ be,
                              float* __restrict__ y, int rows, int Dd) {
  int row = blockIdx.x * blockDim.x + threadIdx.x;
  if (row >= rows) return;
  const float* xp = xin + (size_t)row * Dd;
  const float* rp = r ? r + (size_t)row * Dd : nullptr;
  float mu = 0.f;
  for (int c = 0; c < Dd; ++c) mu += xp[c] + (rp ? rp[c] : 0.f);
  mu /= Dd;
  float var = 0.f;
  for (int c = 0; c < Dd; ++c) {
    float t = xp[c] + (rp ? rp[c] : 0.f) - mu; var += t * t;
  }
  var /= Dd;
  float inv = rsqrtf(var + 1e-5f);
  float* yp = y + (size_t)row * Dd;
  for (int c = 0; c < Dd; ++c)
    yp[c] = (xp[c] + (rp ? rp[c] : 0.f) - mu) * inv * g[c] + be[c];
}

__global__ void gelu_k(float* __restrict__ t, size_t n) {
  size_t tid = blockIdx.x * (size_t)blockDim.x + threadIdx.x;
  if (tid < n) t[tid] = gelu_exact(t[tid]);
}

// broadcast label params across batch
__global__ void bcast_k(const float* __restrict__ src, float* __restrict__ dst, int S) {
  size_t tid = blockIdx.x * (size_t)blockDim.x + threadIdx.x;
  if (tid >= (size_t)BATCH * S * PD) return;
  dst[tid] = src[tid % ((size_t)S * PD)];
}

// per (b,s): k nearest rows of pat (score = |pat_j|^2 - 2 lab_s.pat_j, ordering-equiv)
__global__ void l2p_topk_k(const float* __restrict__ lab, const float* __restrict__ pat,
                           int* __restrict__ idx, int S, int P, int k) {
  int tid = blockIdx.x * blockDim.x + threadIdx.x;
  if (tid >= BATCH * S) return;
  int b = tid / S, s = tid % S;
  const float* lp = lab + ((size_t)b * S + s) * PD;
  const float* pb = pat + (size_t)b * P * PD;
  float bd[9]; int bi[9];
  for (int t = 0; t < k; ++t) { bd[t] = 3.4e38f; bi[t] = 0; }
  for (int j = 0; j < P; ++j) {
    const float* pj = pb + (size_t)j * PD;
    float dot = 0.f, sqj = 0.f;
    for (int c = 0; c < PD; ++c) { float pv = pj[c]; dot += lp[c] * pv; sqj += pv * pv; }
    float d = sqj - 2.f * dot;
    if (d < bd[k - 1]) {
      int pos = k - 1;
      while (pos > 0 && d < bd[pos - 1]) {
        bd[pos] = bd[pos - 1]; bi[pos] = bi[pos - 1]; --pos;
      }
      bd[pos] = d; bi[pos] = j;
    }
  }
  int* op = idx + ((size_t)b * S + s) * k;
  for (int t = 0; t < k; ++t) op[t] = bi[t];
}

// ctx[b,s,c] = max_t pat[b, idx[b,s,t], c] - lab[b,s,c]
__global__ void l2p_gather_k(const float* __restrict__ lab, const float* __restrict__ pat,
                             const int* __restrict__ idx, float* __restrict__ ctx,
                             int S, int P, int k) {
  size_t tid = blockIdx.x * (size_t)blockDim.x + threadIdx.x;
  if (tid >= (size_t)BATCH * S * PD) return;
  int c = tid % PD; size_t r = tid / PD;
  int s = r % S;    int b = (int)(r / S);
  const int* ip = idx + ((size_t)b * S + s) * k;
  float lv = lab[tid];
  float mx = -3.4e38f;
  for (int t = 0; t < k; ++t) {
    float pv = pat[((size_t)b * P + ip[t]) * PD + c];
    mx = fmaxf(mx, pv - lv);
  }
  ctx[tid] = mx;
}

// concat up to 4 (B,S,128) tensors along channels
__global__ void cat4_k(const float* s0, const float* s1, const float* s2, const float* s3,
                       float* __restrict__ dst, int S, int nseg) {
  size_t tid = blockIdx.x * (size_t)blockDim.x + threadIdx.x;
  if (tid >= (size_t)BATCH * S * nseg * PD) return;
  int c = tid % PD; size_t r = tid / PD;
  int seg = r % nseg; r /= nseg;
  int s = r % S;      int b = (int)(r / S);
  const float* src = (seg == 0) ? s0 : (seg == 1) ? s1 : (seg == 2) ? s2 : s3;
  dst[((size_t)b * S + s) * (nseg * PD) + seg * PD + c] = src[((size_t)b * S + s) * PD + c];
}

// C[b,s,n] = sum_k A[b,s,k] * W[n,k] + bias[n]   (small matrices, VALU)
__global__ void linear_small_k(const float* __restrict__ A, const float* __restrict__ Wm,
                               const float* __restrict__ bias, float* __restrict__ C,
                               int S, int N, int Kd) {
  size_t tid = blockIdx.x * (size_t)blockDim.x + threadIdx.x;
  if (tid >= (size_t)BATCH * S * N) return;
  int n = tid % N; size_t r = tid / N;
  int s = r % S;   int b = (int)(r / S);
  const float* ap = A + ((size_t)b * S + s) * Kd;
  const float* wp = Wm + (size_t)n * Kd;
  float acc = bias ? bias[n] : 0.f;
  for (int k2 = 0; k2 < Kd; ++k2) acc += ap[k2] * wp[k2];
  C[tid] = acc;
}

__global__ void pool_k(const float* __restrict__ x, float* __restrict__ pooled) {
  int tid = blockIdx.x * blockDim.x + threadIdx.x;
  if (tid >= BATCH * PD) return;
  int b = tid / PD, c = tid % PD;
  float s = 0.f;
  for (int n = 0; n < NN; ++n) s += x[((size_t)b * NN + n) * PD + c];
  pooled[tid] = s * (1.0f / NN);
}

__global__ void copyL_k(const float* __restrict__ src, float* __restrict__ L,
                        int S, int off) {
  size_t tid = blockIdx.x * (size_t)blockDim.x + threadIdx.x;
  if (tid >= (size_t)BATCH * S * PD) return;
  int c = tid % PD; size_t r = tid / PD;
  int s = r % S;    int b = (int)(r / S);
  L[((size_t)b * 128 + off + s) * PD + c] = src[tid];
}

__global__ void sigmask_k(const float* __restrict__ Araw, const float* __restrict__ mask,
                          float* __restrict__ Asig) {
  int tid = blockIdx.x * blockDim.x + threadIdx.x;
  if (tid >= 128 * 128) return;
  Asig[tid] = (1.0f / (1.0f + expf(-Araw[tid]))) * mask[tid];
}

// out[b,i,c] = sum_j Asig[i,j] * L[b,j,c] + L[b,i,c]
__global__ void aggL_k(const float* __restrict__ Asig, const float* __restrict__ L,
                       float* __restrict__ outb) {
  size_t tid = blockIdx.x * (size_t)blockDim.x + threadIdx.x;
  if (tid >= (size_t)BATCH * 128 * PD) return;
  int c = tid % PD; size_t r = tid / PD;
  int i = r % 128;  int b = (int)(r / 128);
  float acc = L[((size_t)b * 128 + i) * PD + c];
  for (int j = 0; j < 128; ++j)
    acc += Asig[i * 128 + j] * L[((size_t)b * 128 + j) * PD + c];
  outb[tid] = acc;
}

// out[b,i] = sum_c L2[b,i,c] * Wcls[i,c] + ypatch[b,i]
__global__ void yfinal_k(const float* __restrict__ L2, const float* __restrict__ Wcls,
                         const float* __restrict__ ypatch, float* __restrict__ out) {
  int tid = blockIdx.x * blockDim.x + threadIdx.x;
  if (tid >= BATCH * 128) return;
  int b = tid / 128, i = tid % 128;
  float acc = 0.f;
  for (int c = 0; c < PD; ++c)
    acc += L2[((size_t)b * 128 + i) * PD + c] * Wcls[i * PD + c];
  out[tid] = acc + ypatch[tid];
}

// ---------------------------------------------------------------------------
#define L1D(kname, total, ...) \
  kname<<<dim3((unsigned)(((total) + 255) / 256)), dim3(256), 0, stream>>>(__VA_ARGS__)

extern "C" void kernel_launch(void* const* d_in, const int* in_sizes, int n_in,
                              void* d_out, int out_size, void* d_ws, size_t ws_size,
                              hipStream_t stream) {
  if (n_in < 68) return;
  const float* p[68];
  for (int i = 0; i < 68; ++i) p[i] = (const float*)d_in[i];
  // input index map (setup_inputs flattening order):
  // 0 spec; 1..16 convs (w,b,g,be)x4; 17 pos_emb;
  // 18+14*blk: W_in.W,.b, W_out.W,.b, W_upd.W,.b, n1.g,.b, n2.g,.b, ffn1.W,.b, ffn2.W,.b
  // 46 mood, 47 genre, 48 sub, 49/50 W_mood, 51/52 W_genre, 53/54 W_sub,
  // 55/56 nm, 57/58 ng, 59/60 ns, 61 A_raw, 62/63 nL, 64/65 head, 66 W_cls, 67 hmask
  const float* spec = p[0];
  float* out = (float*)d_out;

  // workspace bump allocator (256B aligned)
  char* base = (char*)d_ws;
  size_t off = 0;
  auto alloc = [&](size_t bytes) -> void* {
    off = (off + 255) & ~(size_t)255;
    void* ptr = base + off; off += bytes; return ptr;
  };
  float* P  = (float*)alloc((size_t)33554432 * 4);   // conv1 / conv3 / Gram
  float* Q  = (float*)alloc((size_t)16777216 * 4);   // conv2 / conv4 / t+cat+h+md
  float* x  = (float*)alloc((size_t)BATCH * NN * PD * 4);
  float* h2 = (float*)alloc((size_t)BATCH * NN * PD * 4);
  float* sq = (float*)alloc((size_t)BATCH * NN * 4);
  int* nbr  = (int*)alloc((size_t)BATCH * NN * 9 * 4);
  float* mb = (float*)alloc((size_t)BATCH * 8 * PD * 4);
  float* gb = (float*)alloc((size_t)BATCH * 24 * PD * 4);
  float* sb = (float*)alloc((size_t)BATCH * 96 * PD * 4);
  float* ctx1 = (float*)alloc((size_t)BATCH * 96 * PD * 4);
  float* ctx2 = (float*)alloc((size_t)BATCH * 96 * PD * 4);
  float* ctx3 = (float*)alloc((size_t)BATCH * 96 * PD * 4);
  float* catb = (float*)alloc((size_t)BATCH * 96 * 512 * 4);
  float* linb = (float*)alloc((size_t)BATCH * 96 * PD * 4);
  int* idx    = (int*)alloc((size_t)BATCH * 96 * 9 * 4);
  float* L    = (float*)alloc((size_t)BATCH * 128 * PD * 4);
  float* L2t  = (float*)alloc((size_t)BATCH * 128 * PD * 4);
  float* L2   = (float*)alloc((size_t)BATCH * 128 * PD * 4);
  float* Asig = (float*)alloc((size_t)128 * 128 * 4);
  float* pooled = (float*)alloc((size_t)BATCH * PD * 4);
  float* ypatch = (float*)alloc((size_t)BATCH * PD * 4);
  (void)ws_size; (void)in_sizes; (void)out_size;

  // sub-regions of Q used during PGN (after conv stem is done)
  float* tq   = Q;                         // FFN hidden (B*2048*512)
  float* catq = Q + 8388608;               // [h, md] concat (B*2048*256)
  float* hq   = Q + 12582912;              // h / W_out output (B*2048*128)
  unsigned* mdq = (unsigned*)(Q + 14680064);  // encoded scatter-max (B*2048*128)

  // ---- conv stem ----
  L1D(conv_bn_gelu_k, (size_t)BATCH * 32 * 128 * 1024,
      spec, p[1], p[2], p[3], p[4], P, BATCH, 1, 128, 1024, 32, 128, 1024, 1, 1);
  L1D(conv_bn_gelu_k, (size_t)BATCH * 64 * 64 * 512,
      P, p[5], p[6], p[7], p[8], Q, BATCH, 32, 128, 1024, 64, 64, 512, 2, 0);
  L1D(conv_bn_gelu_k, (size_t)BATCH * 128 * 32 * 256,
      Q, p[9], p[10], p[11], p[12], P, BATCH, 64, 64, 512, 128, 32, 256, 2, 0);
  L1D(conv_bn_gelu_k, (size_t)BATCH * 128 * 16 * 128,
      P, p[13], p[14], p[15], p[16], Q, BATCH, 128, 32, 256, 128, 16, 128, 2, 0);
  L1D(feat_k, (size_t)BATCH * NN * PD, Q, p[17], x);

  // ---- PGN blocks ----
  const dim3 w1(32, 1, 1);
  for (int blk = 0; blk < 2; ++blk) {
    const int bp = 18 + 14 * blk;
    const int dil = blk + 1;
    // Gram matrix G = x . x^T per sample (WMMA, 16x64 strips)
    gemm_xwt_wmma<PD><<<dim3(NN / 64, NN / 16, BATCH), w1, 0, stream>>>(
        x, x, nullptr, P, NN, NN,
        (long long)NN * PD, (long long)NN * PD, (long long)NN * NN);
    L1D(diag_k, BATCH * NN, P, sq);
    L1D(knn_k, BATCH * NN, P, sq, nbr, 9 * dil, dil);
    // h = x . W_in^T + b
    gemm_xwt_wmma<PD><<<dim3(PD / 64, NN / 16, BATCH), w1, 0, stream>>>(
        x, p[bp + 0], p[bp + 1], hq, NN, PD,
        (long long)NN * PD, 0LL, (long long)NN * PD);
    // scatter-max over dst
    L1D(fill_u32_k, (size_t)BATCH * NN * PD, mdq, ENC_NEGINF, (size_t)BATCH * NN * PD);
    L1D(edge_max_k, (size_t)BATCH * NN * 9 * PD, hq, nbr, mdq);
    L1D(concat_hmd_k, (size_t)BATCH * NN * PD, hq, mdq, catq);
    // W_upd (256->128) then W_out (128->128)
    gemm_xwt_wmma<2 * PD><<<dim3(PD / 64, NN / 16, BATCH), w1, 0, stream>>>(
        catq, p[bp + 4], p[bp + 5], h2, NN, PD,
        (long long)NN * 2 * PD, 0LL, (long long)NN * PD);
    gemm_xwt_wmma<PD><<<dim3(PD / 64, NN / 16, BATCH), w1, 0, stream>>>(
        h2, p[bp + 2], p[bp + 3], hq, NN, PD,
        (long long)NN * PD, 0LL, (long long)NN * PD);
    L1D(residual_ln_k, BATCH * NN, x, hq, p[bp + 6], p[bp + 7], x, BATCH * NN, PD);
    // FFN: gelu(x.ffn1^T+b).ffn2^T + b, residual + LN
    gemm_xwt_wmma<PD><<<dim3(512 / 64, NN / 16, BATCH), w1, 0, stream>>>(
        x, p[bp + 10], p[bp + 11], tq, NN, 512,
        (long long)NN * PD, 0LL, (long long)NN * 512);
    L1D(gelu_k, (size_t)BATCH * NN * 512, tq, (size_t)BATCH * NN * 512);
    gemm_xwt_wmma<512><<<dim3(PD / 64, NN / 16, BATCH), w1, 0, stream>>>(
        tq, p[bp + 12], p[bp + 13], h2, NN, PD,
        (long long)NN * 512, 0LL, (long long)NN * PD);
    L1D(residual_ln_k, BATCH * NN, x, h2, p[bp + 8], p[bp + 9], x, BATCH * NN, PD);
  }

  // ---- y_patch: mean pool + head ----
  L1D(pool_k, BATCH * PD, x, pooled);
  L1D(linear_small_k, (size_t)BATCH * 128, pooled, p[64], p[65], ypatch, 1, 128, 128);

  // ---- label path (batched over samples) ----
  L1D(bcast_k, (size_t)BATCH * 8 * PD, p[46], mb, 8);
  L1D(bcast_k, (size_t)BATCH * 24 * PD, p[47], gb, 24);
  L1D(bcast_k, (size_t)BATCH * 96 * PD, p[48], sb, 96);
  // mood
  L1D(l2p_topk_k, BATCH * 8, mb, x, idx, 8, NN, 9);
  L1D(l2p_gather_k, (size_t)BATCH * 8 * PD, mb, x, idx, ctx1, 8, NN, 9);
  L1D(cat4_k, (size_t)BATCH * 8 * 2 * PD, mb, ctx1, nullptr, nullptr, catb, 8, 2);
  L1D(linear_small_k, (size_t)BATCH * 8 * 128, catb, p[49], p[50], linb, 8, 128, 256);
  L1D(residual_ln_k, BATCH * 8, mb, linb, p[55], p[56], mb, BATCH * 8, PD);
  // genre
  L1D(l2p_topk_k, BATCH * 24, gb, x, idx, 24, NN, 9);
  L1D(l2p_gather_k, (size_t)BATCH * 24 * PD, gb, x, idx, ctx1, 24, NN, 9);
  L1D(l2p_topk_k, BATCH * 24, gb, mb, idx, 24, 8, 4);
  L1D(l2p_gather_k, (size_t)BATCH * 24 * PD, gb, mb, idx, ctx2, 24, 8, 4);
  L1D(cat4_k, (size_t)BATCH * 24 * 3 * PD, gb, ctx1, ctx2, nullptr, catb, 24, 3);
  L1D(linear_small_k, (size_t)BATCH * 24 * 128, catb, p[51], p[52], linb, 24, 128, 384);
  L1D(residual_ln_k, BATCH * 24, gb, linb, p[57], p[58], gb, BATCH * 24, PD);
  // sub
  L1D(l2p_topk_k, BATCH * 96, sb, x, idx, 96, NN, 9);
  L1D(l2p_gather_k, (size_t)BATCH * 96 * PD, sb, x, idx, ctx1, 96, NN, 9);
  L1D(l2p_topk_k, BATCH * 96, sb, mb, idx, 96, 8, 3);
  L1D(l2p_gather_k, (size_t)BATCH * 96 * PD, sb, mb, idx, ctx2, 96, 8, 3);
  L1D(l2p_topk_k, BATCH * 96, sb, gb, idx, 96, 24, 4);
  L1D(l2p_gather_k, (size_t)BATCH * 96 * PD, sb, gb, idx, ctx3, 96, 24, 4);
  L1D(cat4_k, (size_t)BATCH * 96 * 4 * PD, sb, ctx1, ctx2, ctx3, catb, 96, 4);
  L1D(linear_small_k, (size_t)BATCH * 96 * 128, catb, p[53], p[54], linb, 96, 128, 512);
  L1D(residual_ln_k, BATCH * 96, sb, linb, p[59], p[60], sb, BATCH * 96, PD);
  // L = [m; g; s]; L2 = LN(sigmoid(A)*mask @ L + L); y_label = sum(L2*W_cls, -1)
  L1D(copyL_k, (size_t)BATCH * 8 * PD, mb, L, 8, 0);
  L1D(copyL_k, (size_t)BATCH * 24 * PD, gb, L, 24, 8);
  L1D(copyL_k, (size_t)BATCH * 96 * PD, sb, L, 96, 32);
  L1D(sigmask_k, 128 * 128, p[61], p[67], Asig);
  L1D(aggL_k, (size_t)BATCH * 128 * PD, Asig, L, L2t);
  L1D(residual_ln_k, BATCH * 128, L2t, nullptr, p[62], p[63], L2, BATCH * 128, PD);
  L1D(yfinal_k, BATCH * 128, L2, p[66], ypatch, out);
}
